// encoderLSTM_15522011807915
// MI455X (gfx1250) — matrix operand order repaired and checked by reading, not compiled, not run on hardware
//
#include <hip/hip_runtime.h>

// ---------------------------------------------------------------------------
// Problem constants (from reference): T=256, N=256, D=512, H=1024
// ---------------------------------------------------------------------------
constexpr int TT = 256;   // timesteps
constexpr int NB = 256;   // batch
constexpr int DD = 512;   // input dim
constexpr int HH = 1024;  // hidden dim

typedef __bf16 bf16;
typedef __bf16 v16bf __attribute__((ext_vector_type(16)));
typedef __bf16 v8bf  __attribute__((ext_vector_type(8)));
typedef float  v8f   __attribute__((ext_vector_type(8)));

// ---------------------------------------------------------------------------
// WMMA fragment helpers (CDNA5 wave32, V_WMMA_F32_16X16X32_BF16)
// A 16x32 (MxK): lane m (0..15), hi = lane>>4.
//   frag[0..7]  = A[m][hi*8 + 0..7]
//   frag[8..15] = A[m][16 + hi*8 + 0..7]
// B 32x16 (KxN) stored as Bt[n][k] row-major:
//   frag[0..7] = Bt[m][hi*8 + 0..7], frag[8..15] = Bt[m][16 + hi*8 + 0..7]
// C/D 16x16 f32: element r -> row (8*hi + r), col m.
// ---------------------------------------------------------------------------
__device__ __forceinline__ v16bf ld_a_frag(const bf16* __restrict__ p, int hi) {
    union { v16bf v; v8bf h[2]; } u;
    u.h[0] = *(const v8bf*)(p + hi * 8);        // K = hi*8 .. hi*8+7
    u.h[1] = *(const v8bf*)(p + 16 + hi * 8);   // K = 16+hi*8 .. 16+hi*8+7
    return u.v;
}

__device__ __forceinline__ v16bf ld_b_frag(const bf16* __restrict__ p, int hi) {
    union { v16bf v; v8bf h[2]; } u;
    u.h[0] = *(const v8bf*)(p + hi * 8);
    u.h[1] = *(const v8bf*)(p + 16 + hi * 8);
    return u.v;
}

__device__ __forceinline__ v8f bf16_mma(v16bf a, v16bf b, v8f c) {
    return __builtin_amdgcn_wmma_f32_16x16x32_bf16(
        /*neg_a=*/false, a, /*neg_b=*/false, b,
        /*c_mod=*/(short)0, c, /*reuse_a=*/false, /*reuse_b=*/false);
}

// ---------------------------------------------------------------------------
// Fast transcendental helpers (lower to v_exp_f32 / v_rcp_f32)
// ---------------------------------------------------------------------------
__device__ __forceinline__ float sigmoid_f(float x) {
    float e = __builtin_amdgcn_exp2f(-x * 1.4426950408889634f); // exp(-x)
    return __builtin_amdgcn_rcpf(1.0f + e);
}
__device__ __forceinline__ float tanh_f(float x) {
    x = fminf(15.0f, fmaxf(-15.0f, x));
    float e = __builtin_amdgcn_exp2f(x * 2.8853900817779268f);  // exp(2x)
    return (e - 1.0f) * __builtin_amdgcn_rcpf(e + 1.0f);
}

// ---------------------------------------------------------------------------
// f32 -> bf16 conversion
// ---------------------------------------------------------------------------
__global__ void cvt_f32_bf16(const float* __restrict__ src, bf16* __restrict__ dst, int n) {
    int i = blockIdx.x * blockDim.x + threadIdx.x;
    if (i < n) dst[i] = (bf16)src[i];
}

// Zero-init cell state (f32) and h0 (bf16)
__global__ void init_state(float* __restrict__ c, bf16* __restrict__ h, int n) {
    int i = blockIdx.x * blockDim.x + threadIdx.x;
    if (i < n) { c[i] = 0.0f; h[i] = (bf16)0.0f; }
}

// ---------------------------------------------------------------------------
// Fused LSTM step.
//   gates[n][g*H+j] = sum_k h_in[n][k]*Whh[g*H+j][k]
//                   + sum_k x_t[n][k] *Wih[g*H+j][k]  + b_ih[g*H+j] + b_hh[g*H+j]
//   i,f,g,o -> c' = f*c + i*g ; h' = o*tanh(c')
// Block tile: 32 batch rows x 32 hidden cols; 4 waves (128 thr); each wave
// owns one 16x16 tile and all 4 gate accumulators for it, so the pointwise
// LSTM update is wave-local. Weights are L2-resident bf16.
// ---------------------------------------------------------------------------
__global__ __launch_bounds__(128)
void lstm_step(const bf16* __restrict__ xt,    // [NB, DD] this timestep
               const bf16* __restrict__ Whh,   // [4H, H] row-major
               const bf16* __restrict__ Wih,   // [4H, D] row-major
               const float* __restrict__ b_ih, // [4H]
               const float* __restrict__ b_hh, // [4H]
               const bf16* __restrict__ h_in,  // [NB, H]
               bf16* __restrict__ h_out,       // [NB, H]
               float* __restrict__ c)          // [NB, H] in-place
{
    const int lane = threadIdx.x & 31;
    const int w    = threadIdx.x >> 5;          // wave 0..3
    const int m    = lane & 15;
    const int hi   = lane >> 4;
    const int row  = blockIdx.x * 32 + (w >> 1) * 16;   // batch base
    const int col  = blockIdx.y * 32 + (w & 1) * 16;    // hidden base

    v8f acc[4] = {v8f{}, v8f{}, v8f{}, v8f{}};          // i, f, g, o

    // --- recurrent part: K = H over h_in ---
    const bf16* aBase = h_in + (size_t)(row + m) * HH;
    for (int k0 = 0; k0 < HH; k0 += 32) {
        __builtin_prefetch(aBase + k0 + 128, 0, 0);     // global_prefetch_b8
        v16bf a = ld_a_frag(aBase + k0, hi);
#pragma unroll
        for (int g = 0; g < 4; ++g) {
            const bf16* bp = Whh + (size_t)(g * HH + col + m) * HH + k0;
            acc[g] = bf16_mma(a, ld_b_frag(bp, hi), acc[g]);
        }
    }

    // --- input part: K = D over x_t ---
    const bf16* xBase = xt + (size_t)(row + m) * DD;
    for (int k0 = 0; k0 < DD; k0 += 32) {
        v16bf a = ld_a_frag(xBase + k0, hi);
#pragma unroll
        for (int g = 0; g < 4; ++g) {
            const bf16* bp = Wih + (size_t)(g * HH + col + m) * DD + k0;
            acc[g] = bf16_mma(a, ld_b_frag(bp, hi), acc[g]);
        }
    }

    // --- pointwise LSTM update ---
    const int j = col + m;
    float bias[4];
#pragma unroll
    for (int g = 0; g < 4; ++g) bias[g] = b_ih[g * HH + j] + b_hh[g * HH + j];

#pragma unroll
    for (int r = 0; r < 8; ++r) {
        const int n = row + 8 * hi + r;
        const size_t idx = (size_t)n * HH + j;
        float gi = sigmoid_f(acc[0][r] + bias[0]);
        float gf = sigmoid_f(acc[1][r] + bias[1]);
        float gg = tanh_f  (acc[2][r] + bias[2]);
        float go = sigmoid_f(acc[3][r] + bias[3]);
        float cn = gf * c[idx] + gi * gg;
        c[idx]     = cn;
        h_out[idx] = (bf16)(go * tanh_f(cn));
    }
}

// ---------------------------------------------------------------------------
// Generic bf16 GEMM with bias:  C[MxN] = A[MxK] @ Bt[NxK]^T + bias[N]
// grid = (M/32, N/32), 128 threads (4 waves), one 16x16 tile per wave.
// Used for the final FC:  out = h @ W_fc^T + b_fc
// ---------------------------------------------------------------------------
__global__ __launch_bounds__(128)
void gemm_bf16_bias(const bf16* __restrict__ A,
                    const bf16* __restrict__ Bt,
                    const float* __restrict__ bias,
                    float* __restrict__ C,
                    int M, int Ncols, int K)
{
    const int lane = threadIdx.x & 31;
    const int w    = threadIdx.x >> 5;
    const int m    = lane & 15;
    const int hi   = lane >> 4;
    const int row  = blockIdx.x * 32 + (w >> 1) * 16;
    const int col  = blockIdx.y * 32 + (w & 1) * 16;

    v8f acc = v8f{};
    const bf16* aBase = A  + (size_t)(row + m) * K;
    const bf16* bBase = Bt + (size_t)(col + m) * K;
    for (int k0 = 0; k0 < K; k0 += 32) {
        v16bf a = ld_a_frag(aBase + k0, hi);
        v16bf b = ld_b_frag(bBase + k0, hi);
        acc = bf16_mma(a, b, acc);
    }

    const float bj = bias ? bias[col + m] : 0.0f;
#pragma unroll
    for (int r = 0; r < 8; ++r) {
        const int n = row + 8 * hi + r;
        C[(size_t)n * Ncols + col + m] = acc[r] + bj;
    }
}

// ---------------------------------------------------------------------------
// Host-side orchestration
// Inputs: x[T,N,D], W_ih[4H,D], W_hh[4H,H], b_ih[4H], b_hh[4H], W_fc[H,H], b_fc[H]
// Output: [N,H] f32
// ---------------------------------------------------------------------------
extern "C" void kernel_launch(void* const* d_in, const int* in_sizes, int n_in,
                              void* d_out, int out_size, void* d_ws, size_t ws_size,
                              hipStream_t stream) {
    const float* x    = (const float*)d_in[0];
    const float* W_ih = (const float*)d_in[1];
    const float* W_hh = (const float*)d_in[2];
    const float* b_ih = (const float*)d_in[3];
    const float* b_hh = (const float*)d_in[4];
    const float* W_fc = (const float*)d_in[5];
    const float* b_fc = (const float*)d_in[6];

    // Workspace layout (all 256B-aligned by construction; ~84 MB total)
    char* ws = (char*)d_ws;
    const size_t nX   = (size_t)TT * NB * DD;      // 33,554,432
    const size_t nWih = (size_t)4 * HH * DD;       //  2,097,152
    const size_t nWhh = (size_t)4 * HH * HH;       //  4,194,304
    const size_t nWfc = (size_t)HH * HH;           //  1,048,576
    const size_t nSt  = (size_t)NB * HH;           //    262,144

    bf16*  x_bf   = (bf16*)ws;                 ws += nX   * sizeof(bf16);
    bf16*  wih_bf = (bf16*)ws;                 ws += nWih * sizeof(bf16);
    bf16*  whh_bf = (bf16*)ws;                 ws += nWhh * sizeof(bf16);
    bf16*  wfc_bf = (bf16*)ws;                 ws += nWfc * sizeof(bf16);
    float* c_st   = (float*)ws;                ws += nSt  * sizeof(float);
    bf16*  h_a    = (bf16*)ws;                 ws += nSt  * sizeof(bf16);
    bf16*  h_b    = (bf16*)ws;                 ws += nSt  * sizeof(bf16);

    // 1) Convert inputs/weights to bf16 (weights then stay L2-resident)
    cvt_f32_bf16<<<(int)((nX   + 255) / 256), 256, 0, stream>>>(x,    x_bf,   (int)nX);
    cvt_f32_bf16<<<(int)((nWih + 255) / 256), 256, 0, stream>>>(W_ih, wih_bf, (int)nWih);
    cvt_f32_bf16<<<(int)((nWhh + 255) / 256), 256, 0, stream>>>(W_hh, whh_bf, (int)nWhh);
    cvt_f32_bf16<<<(int)((nWfc + 255) / 256), 256, 0, stream>>>(W_fc, wfc_bf, (int)nWfc);

    // 2) Zero h0 / c0
    init_state<<<(int)((nSt + 255) / 256), 256, 0, stream>>>(c_st, h_a, (int)nSt);

    // 3) Recurrence: 256 stream-ordered fused steps, double-buffered h
    dim3 stepGrid(NB / 32, HH / 32);   // 8 x 32 = 256 workgroups
    for (int t = 0; t < TT; ++t) {
        lstm_step<<<stepGrid, 128, 0, stream>>>(
            x_bf + (size_t)t * NB * DD, whh_bf, wih_bf, b_ih, b_hh,
            h_a, h_b, c_st);
        bf16* tmp = h_a; h_a = h_b; h_b = tmp;   // final h ends in h_a
    }

    // 4) Final FC: out = h @ W_fc^T + b_fc   -> d_out [NB, HH] f32
    dim3 fcGrid(NB / 32, HH / 32);
    gemm_bf16_bias<<<fcGrid, 128, 0, stream>>>(h_a, wfc_bf, b_fc,
                                               (float*)d_out, NB, HH, HH);
}
